// GraphAttentionLayer_38371237822664
// MI455X (gfx1250) — matrix operand order repaired and checked by reading, not compiled, not run on hardware
//
#include <hip/hip_runtime.h>

typedef __attribute__((ext_vector_type(16))) __bf16 v16bf;
typedef __attribute__((ext_vector_type(8)))  __bf16 v8bf;
typedef __attribute__((ext_vector_type(8)))  float  v8f;
typedef __attribute__((ext_vector_type(2)))  int    v2i;

#define N_NODES 4096
#define IN_FEAT 256
#define N_HEADS 8
#define N_HIDDEN 32
#define OUT_COLS 256   // N_HEADS * N_HIDDEN
#define NEG_INF (-__builtin_inff())

#define AS1 __attribute__((address_space(1)))
#define AS3 __attribute__((address_space(3)))

// global generic pointer -> AS1 typed pointer (addresses are numerically equal)
#define G_I32(p) ((AS1 int*)(unsigned long long)(p))
#define G_V2I(p) ((AS1 v2i*)(unsigned long long)(p))
// LDS generic pointer -> AS3 typed pointer (proper addrspacecast, then retype)
#define L_I32(p) ((AS3 int*)(p))
#define L_V2I(p) ((AS3 v2i*)(AS3 int*)(p))

#if defined(__has_builtin)
#if __has_builtin(__builtin_amdgcn_global_load_async_to_lds_b32) && \
    __has_builtin(__builtin_amdgcn_global_load_async_to_lds_b64) && \
    __has_builtin(__builtin_amdgcn_s_wait_asynccnt)
#define HAVE_ASYNC_LDS 1
#endif
#endif

// ---------------------------------------------------------------------------
// f32 -> bf16 conversion
// ---------------------------------------------------------------------------
__global__ __launch_bounds__(256) void gat_cvt_bf16(const float* __restrict__ src,
                                                    __bf16* __restrict__ dst, int n) {
    int i = blockIdx.x * 256 + threadIdx.x;
    if (i < n) dst[i] = (__bf16)src[i];
}

// ---------------------------------------------------------------------------
// g = h @ W^T  (bf16 WMMA), stored transposed as bf16: gT[col][node]
// one wave computes one 16x16 tile; K=256 in 8 steps of 32
// ---------------------------------------------------------------------------
__global__ __launch_bounds__(256) void gat_gemm(const __bf16* __restrict__ hb,
                                                const __bf16* __restrict__ Wb,
                                                __bf16* __restrict__ gT) {
    const int wave = threadIdx.x >> 5;
    const int lane = threadIdx.x & 31;
    const int hf   = lane >> 4;      // lane half
    const int lm   = lane & 15;
    const int tile = blockIdx.x * 8 + wave;          // 0..4095
    const int m0   = (tile >> 4) << 4;               // node tile
    const int n0   = (tile & 15) << 4;               // feature tile

    v8f c = {};
    // A: rows = nodes (M=lm). half 0 -> K 0-15 ; half 1 -> K 16-31
    const __bf16* pa = hb + (m0 + lm) * IN_FEAT + hf * 16;
    // B: B[k][n] = W[n][k]; lane holds column n = lm; half 0 -> K0-15, half 1 -> K16-31
    const __bf16* pb = Wb + (n0 + lm) * IN_FEAT + hf * 16;

#pragma unroll
    for (int k0 = 0; k0 < IN_FEAT; k0 += 32) {
        v8bf alo = *(const v8bf*)(pa + k0);
        v8bf ahi = *(const v8bf*)(pa + k0 + 8);
        v16bf a;
#pragma unroll
        for (int i = 0; i < 8; ++i) { a[i] = alo[i]; a[8 + i] = ahi[i]; }
        v16bf b = *(const v16bf*)(pb + k0);
        c = __builtin_amdgcn_wmma_f32_16x16x32_bf16(false, a, false, b,
                                                    (short)0, c, false, false);
    }
    // C/D layout: VGPR r -> M = hf*8 + r ; N = lm. Store transposed (col-major).
#pragma unroll
    for (int r = 0; r < 8; ++r) {
        int row = m0 + hf * 8 + r;
        int col = n0 + lm;
        gT[col * N_NODES + row] = (__bf16)c[r];
    }
}

// ---------------------------------------------------------------------------
// s_i[n,h] = sum_f g[n,h,f]*a[f] ; s_j with a[32+f]
// ---------------------------------------------------------------------------
__global__ __launch_bounds__(256) void gat_scores(const __bf16* __restrict__ gT,
                                                  const float* __restrict__ a,
                                                  float* __restrict__ s_i,
                                                  float* __restrict__ s_j) {
    int tid = blockIdx.x * 256 + threadIdx.x;        // 32768 threads
    int n  = tid & (N_NODES - 1);
    int hh = tid >> 12;
    float ai = 0.f, aj = 0.f;
#pragma unroll
    for (int f = 0; f < N_HIDDEN; ++f) {
        float gv = (float)gT[(hh * N_HIDDEN + f) * N_NODES + n];
        ai += gv * a[f];
        aj += gv * a[N_HIDDEN + f];
    }
    s_i[n * N_HEADS + hh] = ai;
    s_j[n * N_HEADS + hh] = aj;
}

// ---------------------------------------------------------------------------
// Fused masked-softmax attention + aggregation (flash-attention style).
// Block = 16-row i-tile; 8 waves = 8 heads; j in tiles of 32 (WMMA K).
// Double-buffered LDS staging of the adj / s_j tiles:
//   - async-to-LDS (ASYNCcnt) path when the gfx1250 builtins exist
//   - register-prefetch + ds_store fallback otherwise
// ---------------------------------------------------------------------------
__global__ __launch_bounds__(256) void gat_attn(const int* __restrict__ adj,
                                                const __bf16* __restrict__ gT,
                                                const float* __restrict__ s_i,
                                                const float* __restrict__ s_j,
                                                float* __restrict__ out) {
    __shared__ int   adjLds[2][16 * 36];   // padded stride 36 dwords: conflict-free
    __shared__ float sTile[2][N_HEADS * 32];

    const int t    = threadIdx.x;
    const int head = t >> 5;
    const int lane = t & 31;
    const int hf   = lane >> 4;
    const int lm   = lane & 15;
    const int i0   = blockIdx.x * 16;

    const float si = s_i[(i0 + lm) * N_HEADS + head];   // my row's left score
    float m = -1e30f, l = 0.0f;
    v8f acc0 = {}, acc1 = {};

    // B fragment base: B[k][n] = g[j0+k, head*32 + n]; gT row = feature, col = node
    const __bf16* pb = gT + (head * N_HIDDEN + lm) * N_NODES + hf * 16;

    // cooperative staging coordinates
    const int ar = t >> 4;               // adj row  0..15
    const int ac = (t & 15) * 2;         // adj cols 0,2..30 (2 ints / thread)
    const int sw = t >> 5;               // s_j head
    const int sk = t & 31;               // s_j col
    const int* aptr_base = adj + (i0 + ar) * N_NODES + ac;   // + j0 per tile

    int buf = 0;

#ifdef HAVE_ASYNC_LDS
    // ---- prologue: async-stage tile 0 into buffer 0 ----
    __builtin_amdgcn_global_load_async_to_lds_b64(
        G_V2I(aptr_base), L_V2I(&adjLds[0][ar * 36 + ac]), 0, 0);
    __builtin_amdgcn_global_load_async_to_lds_b32(
        G_I32(s_j + sk * N_HEADS + sw), L_I32(&sTile[0][sw * 32 + sk]), 0, 0);
#else
    // ---- prologue: prefetch tile 0 into registers ----
    int   ra0 = aptr_base[0];
    int   ra1 = aptr_base[1];
    float rs  = s_j[sk * N_HEADS + sw];
#endif

    for (int j0 = 0; j0 < N_NODES; j0 += 32) {
#ifdef HAVE_ASYNC_LDS
        __builtin_amdgcn_s_wait_asynccnt(0);   // my stage of buf is in LDS
        __syncthreads();                       // everyone's stage is in LDS
        if (j0 + 32 < N_NODES) {               // overlap next tile with compute
            __builtin_amdgcn_global_load_async_to_lds_b64(
                G_V2I(aptr_base + j0 + 32),
                L_V2I(&adjLds[buf ^ 1][ar * 36 + ac]), 0, 0);
            __builtin_amdgcn_global_load_async_to_lds_b32(
                G_I32(s_j + (j0 + 32 + sk) * N_HEADS + sw),
                L_I32(&sTile[buf ^ 1][sw * 32 + sk]), 0, 0);
        }
#else
        adjLds[buf][ar * 36 + ac]     = ra0;
        adjLds[buf][ar * 36 + ac + 1] = ra1;
        sTile[buf][sw * 32 + sk]      = rs;
        __syncthreads();
        if (j0 + 32 < N_NODES) {               // overlap next loads with compute
            ra0 = aptr_base[j0 + 32];
            ra1 = aptr_base[j0 + 33];
            rs  = s_j[(j0 + 32 + sk) * N_HEADS + sw];
        }
#endif
        if (j0 + 64 < N_NODES) __builtin_prefetch(aptr_base + j0 + 64, 0, 1);

        // --- scores for my 16 (row, j) pairs in A-fragment order ---
        float e[16];
        float tmax = NEG_INF;
#pragma unroll
        for (int cdx = 0; cdx < 8; ++cdx) {
            int k  = hf * 16 + cdx;      // K 0-7 / 16-23 per half
            int k2 = k + 8;              // K 8-15 / 24-31
            float v0 = si + sTile[buf][head * 32 + k];
            v0 = (v0 > 0.f) ? v0 : 0.2f * v0;
            if (adjLds[buf][lm * 36 + k] == 0) v0 = NEG_INF;
            float v1 = si + sTile[buf][head * 32 + k2];
            v1 = (v1 > 0.f) ? v1 : 0.2f * v1;
            if (adjLds[buf][lm * 36 + k2] == 0) v1 = NEG_INF;
            e[cdx] = v0; e[8 + cdx] = v1;
            tmax = fmaxf(tmax, fmaxf(v0, v1));
        }
        // row M lives in lanes M and M+16 -> reduce across halves
        tmax = fmaxf(tmax, __shfl_xor(tmax, 16, 32));
        float mn    = fmaxf(m, tmax);
        float scale = __expf(m - mn);

        // rescale accumulators: row (hf*8+r) scale sits in lane (hf*8+r)
#pragma unroll
        for (int r = 0; r < 8; ++r) {
            float sc = __shfl(scale, hf * 8 + r, 32);
            acc0[r] *= sc;
            acc1[r] *= sc;
        }

        // p = exp(e - mn), packed to bf16 A fragment; row sum
        v16bf aP;
        float psum = 0.f;
#pragma unroll
        for (int i = 0; i < 16; ++i) {
            float p = __expf(e[i] - mn);
            aP[i] = (__bf16)p;
            psum += p;
        }
        psum += __shfl_xor(psum, 16, 32);
        l = l * scale + psum;
        m = mn;

        // B fragments: features 0-15 and 16-31, K contiguous in gT
        v16bf b0 = *(const v16bf*)(pb + j0);
        v16bf b1 = *(const v16bf*)(pb + 16 * N_NODES + j0);
        acc0 = __builtin_amdgcn_wmma_f32_16x16x32_bf16(false, aP, false, b0,
                                                       (short)0, acc0, false, false);
        acc1 = __builtin_amdgcn_wmma_f32_16x16x32_bf16(false, aP, false, b1,
                                                       (short)0, acc1, false, false);
        buf ^= 1;
    }

    float inv = 1.0f / l;   // per row, valid in lanes lm and lm+16
#pragma unroll
    for (int r = 0; r < 8; ++r) {
        float iv = __shfl(inv, hf * 8 + r, 32);
        int row = i0 + hf * 8 + r;
        out[row * OUT_COLS + head * N_HIDDEN + lm]      = acc0[r] * iv;
        out[row * OUT_COLS + head * N_HIDDEN + 16 + lm] = acc1[r] * iv;
    }
}

// ---------------------------------------------------------------------------
extern "C" void kernel_launch(void* const* d_in, const int* in_sizes, int n_in,
                              void* d_out, int out_size, void* d_ws, size_t ws_size,
                              hipStream_t stream) {
    const float* h   = (const float*)d_in[0];
    const float* W   = (const float*)d_in[1];
    const float* a   = (const float*)d_in[2];
    const int*   adj = (const int*)d_in[3];
    float* out = (float*)d_out;

    char* ws = (char*)d_ws;
    __bf16* h_bf = (__bf16*)(ws);                      // 4096*256*2 = 2 MB
    __bf16* W_bf = (__bf16*)(ws + (2u << 20));         // 256*256*2  = 128 KB
    __bf16* gT   = (__bf16*)(ws + (2u << 20) + (128u << 10));  // 2 MB
    float*  si   = (float*)(ws + (4u << 20) + (256u << 10));   // 128 KB
    float*  sj   = (float*)(ws + (4u << 20) + (384u << 10));   // 128 KB

    gat_cvt_bf16<<<(N_NODES * IN_FEAT) / 256, 256, 0, stream>>>(h, h_bf, N_NODES * IN_FEAT);
    gat_cvt_bf16<<<(OUT_COLS * IN_FEAT) / 256, 256, 0, stream>>>(W, W_bf, OUT_COLS * IN_FEAT);
    gat_gemm<<<(N_NODES / 16) * (OUT_COLS / 16) / 8, 256, 0, stream>>>(h_bf, W_bf, gT);
    gat_scores<<<(N_NODES * N_HEADS) / 256, 256, 0, stream>>>(gT, a, si, sj);
    gat_attn<<<N_NODES / 16, 256, 0, stream>>>(adj, gT, si, sj, out);
}
